// MultiHeadAttention_32770600469026
// MI455X (gfx1250) — matrix operand order repaired and checked by reading, not compiled, not run on hardware
//
#include <hip/hip_runtime.h>
#include <hip/hip_bf16.h>

// CDNA5 (gfx1250) wave32 WMMA implementation of fused MHA block.
// B=2, L=4096, D=768, H=12, HD=64.

typedef __bf16 bf16_t;
typedef __attribute__((ext_vector_type(16))) __bf16 v16bf;
typedef __attribute__((ext_vector_type(8)))  __bf16 v8bf;   // 16 bytes
typedef __attribute__((ext_vector_type(8)))  float  v8f;
typedef __attribute__((ext_vector_type(4)))  int    v4i;    // 16 bytes

// AS-qualified vector pointees for the async-LDS builtin
// (param 1 printed by clang as "v4i __device__ *"  == AS(1);
//  param 2 is the LDS destination               == AS(3)).
typedef __attribute__((address_space(1))) v4i gv4i;
typedef __attribute__((address_space(3))) v4i lv4i;

#define WMMA_BF16(a, b, c) \
  __builtin_amdgcn_wmma_f32_16x16x32_bf16(false, (a), false, (b), (short)0, (c), false, false)

#if defined(__gfx1250__) && __has_builtin(__builtin_amdgcn_global_load_async_to_lds_b128)
#define HAVE_ASYNC_LDS 1
#else
#define HAVE_ASYNC_LDS 0
#endif

// 16-byte global -> LDS copy; async (ASYNCcnt-tracked, no VGPR bounce) if available.
__device__ __forceinline__ void copy16_g2lds(const bf16_t* gsrc, bf16_t* ldst) {
#if HAVE_ASYNC_LDS
  __builtin_amdgcn_global_load_async_to_lds_b128((gv4i*)gsrc, (lv4i*)ldst, 0, 0);
#else
  *(uint4*)ldst = *(const uint4*)gsrc;
#endif
}

__device__ __forceinline__ void wait_g2lds() {
#if HAVE_ASYNC_LDS
#if __has_builtin(__builtin_amdgcn_s_wait_asynccnt)
  __builtin_amdgcn_s_wait_asynccnt(0);
#else
  asm volatile("s_wait_asynccnt 0" ::: "memory");
#endif
#endif
}

__device__ __forceinline__ v8f zero_v8() {
  v8f z;
#pragma unroll
  for (int i = 0; i < 8; ++i) z[i] = 0.0f;
  return z;
}

#define CAT16 0,1,2,3,4,5,6,7,8,9,10,11,12,13,14,15

// A-fragment (16x32 bf16, row-major, stride lda elems; lda*2 must be 16B-aligned).
// ISA layout: lanes 0-15 row M=lane, elems 0..7 -> K 0..7, 8..15 -> K 16..23;
// lanes 16-31: K 8..15 / 24..31.  Two b128 loads per lane.
__device__ __forceinline__ v16bf load_a_frag(const bf16_t* __restrict__ A, int lda, int lane) {
  const int m  = lane & 15;
  const int kh = (lane >> 4) * 8;
  const v8bf lo = *(const v8bf*)(A + m * lda + kh);
  const v8bf hi = *(const v8bf*)(A + m * lda + 16 + kh);
  return __builtin_shufflevector(lo, hi, CAT16);
}

// B-fragment for math matrix Bmat[k][n] stored transposed: Bmat[k][n] = S[(n0+n)*lds + kofs + k].
// Per-lane reads: K 8h..8h+7 and 16+8h..16+8h+7 (same order as A) -> two b128 loads.
__device__ __forceinline__ v16bf load_bT_frag(const bf16_t* __restrict__ S, int lds, int n0,
                                              int kofs, int lane) {
  const int n  = n0 + (lane & 15);
  const int kb = kofs + ((lane >> 4) << 3);
  const v8bf lo = *(const v8bf*)(S + n * lds + kb);
  const v8bf hi = *(const v8bf*)(S + n * lds + kb + 16);
  return __builtin_shufflevector(lo, hi, CAT16);
}

// ---------------------------------------------------------------------------
// Kernel 0: fp32 -> bf16 conversion
// ---------------------------------------------------------------------------
__global__ void k_f32_to_bf16(const float* __restrict__ in, bf16_t* __restrict__ out, int n) {
  int i = blockIdx.x * blockDim.x + threadIdx.x;
  if (i < n) out[i] = (bf16_t)in[i];
}

// ---------------------------------------------------------------------------
// Kernel 1/4: GEMM  C[M,N] = A[M,K] * W^T   (W stored [N,K] row-major)
// 64x64 C-tile per block, 128 threads = 4 waves, wave w owns rows [w*16, w*16+16).
// Both tiles staged in natural [row][k] layout so all frag reads are b128.
// ---------------------------------------------------------------------------
__global__ __launch_bounds__(128)
void k_gemm_bf16(const bf16_t* __restrict__ A, const bf16_t* __restrict__ W,
                 float* __restrict__ C, int M, int N, int K) {
  __shared__ bf16_t As[64 * 40];  // 64 rows x 32 K, stride 40 (80B, 16B-aligned)
  __shared__ bf16_t Ws[64 * 40];  // 64 N-rows x 32 K
  const int tid  = threadIdx.x;
  const int lane = tid & 31;
  const int wave = tid >> 5;
  const int nTiles = N >> 6;
  const int bm = (blockIdx.x / nTiles) * 64;
  const int bn = (blockIdx.x % nTiles) * 64;

  v8f acc[4];
#pragma unroll
  for (int t = 0; t < 4; ++t) acc[t] = zero_v8();

  for (int kk = 0; kk < K; kk += 32) {
    // 64x32 tiles = 256 uint4 each; 2 per thread per tile.
#pragma unroll
    for (int i = 0; i < 2; ++i) {
      int t = tid + i * 128;
      int r = t >> 2, kc = (t & 3) * 8;
      copy16_g2lds(&A[(size_t)(bm + r) * K + kk + kc], &As[r * 40 + kc]);
      copy16_g2lds(&W[(size_t)(bn + r) * K + kk + kc], &Ws[r * 40 + kc]);
    }
    wait_g2lds();
    __syncthreads();
    v16bf a = load_a_frag(&As[wave * 16 * 40], 40, lane);
#pragma unroll
    for (int nt = 0; nt < 4; ++nt) {
      v16bf b = load_bT_frag(Ws, 40, nt * 16, 0, lane);
      acc[nt] = WMMA_BF16(a, b, acc[nt]);
    }
    __syncthreads();
  }

  const int hi = lane >> 4, lo = lane & 15;
#pragma unroll
  for (int nt = 0; nt < 4; ++nt)
#pragma unroll
    for (int r = 0; r < 8; ++r)
      C[(size_t)(bm + wave * 16 + r + 8 * hi) * N + bn + nt * 16 + lo] = acc[nt][r];
}

// ---------------------------------------------------------------------------
// Kernel 2: per-(b,l,h) RMSNorm + RoPE.  Q,K -> [B,H,L,64] bf16 (Q pre-scaled
// by 1/sqrt(64)); V -> dim-major [B,H,64,L] bf16 so attention V-frags are
// contiguous in LDS.
// ---------------------------------------------------------------------------
__global__ __launch_bounds__(64)
void k_norm_rope(const float* __restrict__ qkv, const float* __restrict__ cosT,
                 const float* __restrict__ sinT, const float* __restrict__ qw,
                 const float* __restrict__ kw, bf16_t* __restrict__ Qb,
                 bf16_t* __restrict__ Kb, bf16_t* __restrict__ Vt) {
  __shared__ float red[2][64];
  __shared__ float qs[64], ks[64];
  const int idx = blockIdx.x;
  const int h = idx % 12;
  const int l = (idx / 12) % 4096;
  const int b = idx / (12 * 4096);
  const int hd = threadIdx.x;

  const size_t row = (size_t)(b * 4096 + l) * 2304;
  float q = qkv[row + h * 64 + hd];
  float k = qkv[row + 768 + h * 64 + hd];
  float v = qkv[row + 1536 + h * 64 + hd];

  red[0][hd] = q * q;
  red[1][hd] = k * k;
  __syncthreads();
  for (int s = 32; s > 0; s >>= 1) {
    if (hd < s) { red[0][hd] += red[0][hd + s]; red[1][hd] += red[1][hd + s]; }
    __syncthreads();
  }
  float qn = q * rsqrtf(red[0][0] * (1.0f / 64.0f) + 1e-6f) * qw[hd];
  float kn = k * rsqrtf(red[1][0] * (1.0f / 64.0f) + 1e-6f) * kw[hd];
  qs[hd] = qn; ks[hd] = kn;
  __syncthreads();

  float c = cosT[l * 64 + hd], s = sinT[l * 64 + hd];
  float qr, kr;
  if (hd < 32) { qr = qn * c - qs[hd + 32] * s; kr = kn * c - ks[hd + 32] * s; }
  else         { qr = qn * c + qs[hd - 32] * s; kr = kn * c + ks[hd - 32] * s; }

  const size_t o = ((size_t)(b * 12 + h) * 4096 + l) * 64 + hd;
  Qb[o] = (bf16_t)(qr * 0.125f);
  Kb[o] = (bf16_t)kr;
  Vt[((size_t)(b * 12 + h) * 64 + hd) * 4096 + l] = (bf16_t)v;   // dim-major
}

// ---------------------------------------------------------------------------
// Kernel 3: flash attention. Block = 128 thr (4 waves); wave owns 16 Q rows.
// 32-key K/V tiles streamed through LDS; online softmax; out bf16 [B*L, 768].
// ---------------------------------------------------------------------------
__global__ __launch_bounds__(128)
void k_attn(const bf16_t* __restrict__ Qb, const bf16_t* __restrict__ Kb,
            const bf16_t* __restrict__ Vt, bf16_t* __restrict__ Ob) {
  __shared__ bf16_t Ks[32 * 72];      // [key][dim], stride 72 (144B, 16B-aligned)
  __shared__ bf16_t Vts[64 * 40];     // [dim][key], stride 40
  __shared__ bf16_t Ps[4][16 * 40];   // per-wave P scratch, stride 40

  const int tid = threadIdx.x, lane = tid & 31, wave = tid >> 5;
  const int qt = blockIdx.x, h = blockIdx.y, b = blockIdx.z;
  const size_t base  = (size_t)(b * 12 + h) * 4096 * 64;   // [B,H,L,64] q/k
  const size_t vbase = (size_t)(b * 12 + h) * 64 * 4096;   // [B,H,64,L] v
  const int q0 = qt * 64 + wave * 16;
  const int hi = lane >> 4, lo = lane & 15;

  // Q fragments in registers for the whole loop (K=64 -> 2 A-frags).
  const v8bf* qv = (const v8bf*)(Qb + base + (size_t)(q0 + lo) * 64);
  v16bf a0 = __builtin_shufflevector(qv[hi],     qv[2 + hi], CAT16);
  v16bf a1 = __builtin_shufflevector(qv[4 + hi], qv[6 + hi], CAT16);

  float mrow[8], lrow[8];
  v8f o[4];
#pragma unroll
  for (int r = 0; r < 8; ++r) { mrow[r] = -1e30f; lrow[r] = 0.0f; }
#pragma unroll
  for (int nt = 0; nt < 4; ++nt) o[nt] = zero_v8();

  for (int jt = 0; jt < 4096; jt += 32) {
    __syncthreads();
#pragma unroll
    for (int i = 0; i < 2; ++i) {
      int t = tid + i * 128;
      { // K tile: [key][dim]  (256 uint4)
        int kr = t >> 3, dc = (t & 7) * 8;
        copy16_g2lds(&Kb[base + (size_t)(jt + kr) * 64 + dc], &Ks[kr * 72 + dc]);
      }
      { // V tile: [dim][key]  (256 uint4)
        int d = t >> 2, kc = (t & 3) * 8;
        copy16_g2lds(&Vt[vbase + (size_t)d * 4096 + jt + kc], &Vts[d * 40 + kc]);
      }
    }
    if (jt + 32 < 4096) {  // prefetch next K tile -> global_prefetch_b8
      __builtin_prefetch(&Kb[base + (size_t)(jt + 32 + (tid >> 2)) * 64], 0, 1);
    }
    wait_g2lds();
    __syncthreads();

    // S = Q * K^T  (16 q-rows x 32 keys); Q pre-scaled by 1/sqrt(HD).
    v8f s0 = WMMA_BF16(a0, load_bT_frag(Ks, 72, 0, 0, lane), zero_v8());
    s0     = WMMA_BF16(a1, load_bT_frag(Ks, 72, 0, 32, lane), s0);
    v8f s1 = WMMA_BF16(a0, load_bT_frag(Ks, 72, 16, 0, lane), zero_v8());
    s1     = WMMA_BF16(a1, load_bT_frag(Ks, 72, 16, 32, lane), s1);

    // Online softmax: row r+8*hi lives across the 16 lanes sharing hi.
#pragma unroll
    for (int r = 0; r < 8; ++r) {
      float vmax = fmaxf(s0[r], s1[r]);
#pragma unroll
      for (int msk = 1; msk < 16; msk <<= 1) vmax = fmaxf(vmax, __shfl_xor(vmax, msk, 32));
      float mnew = fmaxf(mrow[r], vmax);
      float p0 = __expf(s0[r] - mnew);
      float p1 = __expf(s1[r] - mnew);
      float psum = p0 + p1;
#pragma unroll
      for (int msk = 1; msk < 16; msk <<= 1) psum += __shfl_xor(psum, msk, 32);
      float alpha = __expf(mrow[r] - mnew);
      mrow[r] = mnew;
      lrow[r] = lrow[r] * alpha + psum;
#pragma unroll
      for (int nt = 0; nt < 4; ++nt) o[nt][r] *= alpha;
      Ps[wave][(r + 8 * hi) * 40 + lo]      = (bf16_t)p0;
      Ps[wave][(r + 8 * hi) * 40 + 16 + lo] = (bf16_t)p1;
    }
    // P round-trips LDS (per-wave region; wave32 DS ops are in-order, wait for stores).
    asm volatile("s_wait_dscnt 0" ::: "memory");
    v16bf ap = load_a_frag(&Ps[wave][0], 40, lane);
#pragma unroll
    for (int nt = 0; nt < 4; ++nt)
      o[nt] = WMMA_BF16(ap, load_bT_frag(Vts, 40, nt * 16, 0, lane), o[nt]);
  }

  // Finalize: divide by l, write bf16 [B*L, 768] for the output projection.
#pragma unroll
  for (int r = 0; r < 8; ++r) {
    float inv = 1.0f / lrow[r];
    int l = q0 + r + 8 * hi;
    size_t orow = ((size_t)b * 4096 + l) * 768 + h * 64;
#pragma unroll
    for (int nt = 0; nt < 4; ++nt)
      Ob[orow + nt * 16 + lo] = (bf16_t)(o[nt][r] * inv);
  }
}

// ---------------------------------------------------------------------------
extern "C" void kernel_launch(void* const* d_in, const int* in_sizes, int n_in,
                              void* d_out, int out_size, void* d_ws, size_t ws_size,
                              hipStream_t stream) {
  (void)in_sizes; (void)n_in; (void)out_size; (void)ws_size;
  const float* x    = (const float*)d_in[0];
  const float* rc   = (const float*)d_in[1];
  const float* rs   = (const float*)d_in[2];
  const float* wqkv = (const float*)d_in[3];
  const float* wout = (const float*)d_in[4];
  const float* qw   = (const float*)d_in[5];
  const float* kw   = (const float*)d_in[6];
  float* out = (float*)d_out;

  char* ws = (char*)d_ws;
  auto alloc = [&](size_t bytes) -> char* {
    char* p = ws;
    ws += (bytes + 255) & ~(size_t)255;
    return p;
  };
  const size_t NX = (size_t)2 * 4096 * 768;          // B*L*D
  bf16_t* xb    = (bf16_t*)alloc(NX * 2);
  bf16_t* wqkvb = (bf16_t*)alloc((size_t)2304 * 768 * 2);
  bf16_t* woutb = (bf16_t*)alloc((size_t)768 * 768 * 2);
  float*  qkv   = (float*) alloc((size_t)8192 * 2304 * 4);
  bf16_t* Qb    = (bf16_t*)alloc(NX * 2);
  bf16_t* Kb    = (bf16_t*)alloc(NX * 2);
  bf16_t* Vt    = (bf16_t*)alloc(NX * 2);
  bf16_t* Ob    = (bf16_t*)alloc(NX * 2);

  k_f32_to_bf16<<<(int)((NX + 255) / 256), 256, 0, stream>>>(x, xb, (int)NX);
  k_f32_to_bf16<<<(2304 * 768 + 255) / 256, 256, 0, stream>>>(wqkv, wqkvb, 2304 * 768);
  k_f32_to_bf16<<<(768 * 768 + 255) / 256, 256, 0, stream>>>(wout, woutb, 768 * 768);

  // QKV projection: [8192,768] x [768,2304]
  k_gemm_bf16<<<128 * 36, 128, 0, stream>>>(xb, wqkvb, qkv, 8192, 2304, 768);

  // RMSNorm + RoPE (+ V transpose to dim-major)
  k_norm_rope<<<2 * 4096 * 12, 64, 0, stream>>>(qkv, rc, rs, qw, kw, Qb, Kb, Vt);

  // Attention
  dim3 g(64, 12, 2);
  k_attn<<<g, 128, 0, stream>>>(Qb, Kb, Vt, Ob);

  // Output projection -> fp32 d_out
  k_gemm_bf16<<<128 * 12, 128, 0, stream>>>(Ob, woutb, out, 8192, 768, 768);
}
